// GraphProcessor_42898133352985
// MI455X (gfx1250) — compile-verified, matrix operand order checked by reading, not run on hardware
//
#include <hip/hip_runtime.h>
#include <hip/hip_bf16.h>

#define DF 128

typedef __attribute__((ext_vector_type(16))) __bf16 v16bf;
typedef __attribute__((ext_vector_type(8)))  float  v8f;

struct FragU { uint4 a, b; };

__device__ __forceinline__ unsigned short f2bf(float f) {
  unsigned u = __float_as_uint(f);
  u += 0x7FFFu + ((u >> 16) & 1u);          // round-to-nearest-even
  return (unsigned short)(u >> 16);
}
__device__ __forceinline__ unsigned pk2(float a, float b) {
  return (unsigned)f2bf(a) | ((unsigned)f2bf(b) << 16);
}
__device__ __forceinline__ uint4 pk8(float4 a, float4 b) {
  return make_uint4(pk2(a.x, a.y), pk2(a.z, a.w), pk2(b.x, b.y), pk2(b.z, b.w));
}
__device__ __forceinline__ float4 ld4(const float* p) { return *(const float4*)p; }

__device__ __forceinline__ void wait_lds() {
  asm volatile("s_wait_dscnt 0" ::: "memory");
}

// 16x32 bf16 fragment from LDS tile with padded row stride (stride % 8 == 0 elems,
// stride*2 % 256 == 16 bytes -> rows start at distinct bank groups).
__device__ __forceinline__ v16bf load_frag(const unsigned short* base, int stride, int lane) {
  const int m = lane & 15, hl = lane >> 4;
  const unsigned short* p = base + m * stride + hl * 8;
  FragU f;
  f.a = *(const uint4*)p;         // K = hl*8 .. +7
  f.b = *(const uint4*)(p + 16);  // K = 16 + hl*8 .. +7
  return __builtin_bit_cast(v16bf, f);
}

__device__ __forceinline__ void zero_acc(v8f acc[8]) {
#pragma unroll
  for (int i = 0; i < 8; ++i)
#pragma unroll
    for (int j = 0; j < 8; ++j) acc[i][j] = 0.f;
}

// [32 x K] tile (LDS bf16, padded stride KP) @ [K x 128] (LDS bf16, transposed, stride KP).
// Two M sub-tiles share every B fragment: 2 WMMAs per B load -> ~160 B/clk LDS demand.
template <int K, int KP>
__device__ __forceinline__ void gemm_tile2(const unsigned short* A, const unsigned short* W,
                                           int lane, v8f acc0[8], v8f acc1[8]) {
  for (int kc = 0; kc < K / 32; ++kc) {
    v16bf a0 = load_frag(A + kc * 32, KP, lane);
    v16bf a1 = load_frag(A + 16 * KP + kc * 32, KP, lane);
#pragma unroll
    for (int nt = 0; nt < 8; ++nt) {
      v16bf b = load_frag(W + nt * 16 * KP + kc * 32, KP, lane);
      acc0[nt] = __builtin_amdgcn_wmma_f32_16x16x32_bf16(
          false, a0, false, b, (short)0, acc0[nt], false, false);
      acc1[nt] = __builtin_amdgcn_wmma_f32_16x16x32_bf16(
          false, a1, false, b, (short)0, acc1[nt], false, false);
    }
  }
}

// C/D layout: element (m,n): vgpr r=m&7, lane = (m>>3)*16 + n
__device__ __forceinline__ void bias_relu_store2(const v8f acc0[8], const v8f acc1[8],
                                                 const float* __restrict__ b,
                                                 unsigned short* dst, int stride, int lane) {
  const int nlo = lane & 15, hl = lane >> 4;
#pragma unroll
  for (int nt = 0; nt < 8; ++nt) {
    const int n = nt * 16 + nlo;
    const float bb = b[n];
#pragma unroll
    for (int r = 0; r < 8; ++r) {
      dst[(r + 8 * hl) * stride + n]      = f2bf(fmaxf(acc0[nt][r] + bb, 0.f));
      dst[(16 + r + 8 * hl) * stride + n] = f2bf(fmaxf(acc1[nt][r] + bb, 0.f));
    }
  }
}

__device__ __forceinline__ void bias_store2_f32(const v8f acc0[8], const v8f acc1[8],
                                                const float* __restrict__ b,
                                                float* dst, int stride, int lane) {
  const int nlo = lane & 15, hl = lane >> 4;
#pragma unroll
  for (int nt = 0; nt < 8; ++nt) {
    const int n = nt * 16 + nlo;
    const float bb = b[n];
#pragma unroll
    for (int r = 0; r < 8; ++r) {
      dst[(r + 8 * hl) * stride + n]      = acc0[nt][r] + bb;
      dst[(16 + r + 8 * hl) * stride + n] = acc1[nt][r] + bb;
    }
  }
}

// LDS layout constants (bf16 elems unless noted)
#define EW_LDS 84992   // eW1T(128x392) + eW2T(128x136) + eW3T(128x136)
#define NW_LDS 68608   // nW1T(128x264) + nW2T(128x136) + nW3T(128x136)
#define WV_BYTES 34304 // per-wave staging region (32-row tiles)

// ---------------- Edge update: ea += LN(MLP([x[row], x[col], ea])), fused scatter into agg
__global__ __launch_bounds__(128) void edge_kernel(
    const float* __restrict__ x, const int* __restrict__ rows, const int* __restrict__ cols,
    float* __restrict__ ea, float* __restrict__ agg,
    const unsigned short* __restrict__ Wp,
    const float* __restrict__ b1, const float* __restrict__ b2, const float* __restrict__ b3,
    const float* __restrict__ gg, const float* __restrict__ bt, int E) {
  extern __shared__ char smem[];
  unsigned short* sW1 = (unsigned short*)smem;   // 128 x 392
  unsigned short* sW2 = sW1 + 50176;             // 128 x 136
  unsigned short* sW3 = sW1 + 67584;             // 128 x 136
  const int wave = threadIdx.x >> 5, lane = threadIdx.x & 31;
  char* wb = smem + 2 * EW_LDS + wave * WV_BYTES;
  unsigned short* sA1 = (unsigned short*)wb;            // 32 x 392 (layer-1 input)
  unsigned short* sA3 = (unsigned short*)wb;            // 32 x 136 (overlay, A1 dead)
  float*          sLN = (float*)(wb + 8704);            // 32 x 132 f32 (overlay)
  unsigned short* sA2 = (unsigned short*)(wb + 25600);  // 32 x 136

  {  // stage padded, pre-transposed bf16 weights (contiguous copy)
    const uint4* src = (const uint4*)Wp;
    uint4* dst = (uint4*)sW1;
    for (int i = threadIdx.x; i < 2 * EW_LDS / 16; i += blockDim.x) dst[i] = src[i];
  }
  __syncthreads();

  const int nTiles = (E + 31) >> 5;
  const int wstride = gridDim.x * (blockDim.x >> 5);
  for (int tile = blockIdx.x * (blockDim.x >> 5) + wave; tile < nTiles; tile += wstride) {
    const int e = (tile << 5) + lane;   // one lane per edge row
    if (tile + wstride < nTiles) {
      const int ep = ((tile + wstride) << 5) + lane;
      if (ep < E) __builtin_prefetch(ea + (size_t)ep * DF, 0, 0);
    }
    if (e < E) {  // stage A1 = [x[src], x[dst], ea] as bf16
      const float* sS = x + (size_t)rows[e] * DF;
      const float* sD = x + (size_t)cols[e] * DF;
      const float* sE = ea + (size_t)e * DF;
      unsigned short* dst = sA1 + lane * 392;
#pragma unroll
      for (int c = 0; c < 128; c += 8) {
        *(uint4*)(dst + c)       = pk8(ld4(sS + c), ld4(sS + c + 4));
        *(uint4*)(dst + 128 + c) = pk8(ld4(sD + c), ld4(sD + c + 4));
        *(uint4*)(dst + 256 + c) = pk8(ld4(sE + c), ld4(sE + c + 4));
      }
    }
    wait_lds();

    v8f acc0[8], acc1[8];
    zero_acc(acc0); zero_acc(acc1);
    gemm_tile2<384, 392>(sA1, sW1, lane, acc0, acc1);
    bias_relu_store2(acc0, acc1, b1, sA2, 136, lane);
    wait_lds();
    zero_acc(acc0); zero_acc(acc1);
    gemm_tile2<128, 136>(sA2, sW2, lane, acc0, acc1);
    bias_relu_store2(acc0, acc1, b2, sA3, 136, lane);
    wait_lds();
    zero_acc(acc0); zero_acc(acc1);
    gemm_tile2<128, 136>(sA3, sW3, lane, acc0, acc1);
    bias_store2_f32(acc0, acc1, b3, sLN, 132, lane);
    wait_lds();

    if (e < E) {  // LayerNorm + residual + fused scatter-add; one lane per row
      const float* hrow = sLN + lane * 132;
      float s = 0.f;
      for (int j = 0; j < 128; j += 4) {
        float4 h = *(const float4*)(hrow + j);
        s += h.x + h.y + h.z + h.w;
      }
      const float mu = s * (1.0f / 128.0f);
      float vs = 0.f;
      for (int j = 0; j < 128; j += 4) {
        float4 h = *(const float4*)(hrow + j);
        float a0 = h.x - mu, a1 = h.y - mu, a2 = h.z - mu, a3 = h.w - mu;
        vs += a0 * a0 + a1 * a1 + a2 * a2 + a3 * a3;
      }
      const float rstd = rsqrtf(vs * (1.0f / 128.0f) + 1e-5f);
      float* eaRow = ea + (size_t)e * DF;
      float* aggRow = agg + (size_t)cols[e] * DF;
      for (int j = 0; j < 128; j += 4) {
        float4 h = *(const float4*)(hrow + j);
        float4 g4 = ld4(gg + j);
        float4 b4 = ld4(bt + j);
        float4 eo = ld4(eaRow + j);
        float4 o;
        o.x = eo.x + (h.x - mu) * rstd * g4.x + b4.x;
        o.y = eo.y + (h.y - mu) * rstd * g4.y + b4.y;
        o.z = eo.z + (h.z - mu) * rstd * g4.z + b4.z;
        o.w = eo.w + (h.w - mu) * rstd * g4.w + b4.w;
        *(float4*)(eaRow + j) = o;
        atomicAdd(aggRow + j + 0, o.x);
        atomicAdd(aggRow + j + 1, o.y);
        atomicAdd(aggRow + j + 2, o.z);
        atomicAdd(aggRow + j + 3, o.w);
      }
    }
  }
}

// ---------------- Node update: x += LN(MLP([x, agg]))
__global__ __launch_bounds__(128) void node_kernel(
    float* __restrict__ x, const float* __restrict__ agg,
    const unsigned short* __restrict__ Wp,
    const float* __restrict__ b1, const float* __restrict__ b2, const float* __restrict__ b3,
    const float* __restrict__ gg, const float* __restrict__ bt, int N) {
  extern __shared__ char smem[];
  unsigned short* sW1 = (unsigned short*)smem;   // 128 x 264
  unsigned short* sW2 = sW1 + 33792;             // 128 x 136
  unsigned short* sW3 = sW1 + 51200;             // 128 x 136
  const int wave = threadIdx.x >> 5, lane = threadIdx.x & 31;
  char* wb = smem + 2 * NW_LDS + wave * WV_BYTES;
  unsigned short* sA1 = (unsigned short*)wb;            // 32 x 264
  unsigned short* sA3 = (unsigned short*)wb;            // 32 x 136 (overlay)
  float*          sLN = (float*)(wb + 8704);            // 32 x 132 f32 (overlay)
  unsigned short* sA2 = (unsigned short*)(wb + 25600);  // 32 x 136

  {
    const uint4* src = (const uint4*)Wp;
    uint4* dst = (uint4*)sW1;
    for (int i = threadIdx.x; i < 2 * NW_LDS / 16; i += blockDim.x) dst[i] = src[i];
  }
  __syncthreads();

  const int nTiles = (N + 31) >> 5;
  const int wstride = gridDim.x * (blockDim.x >> 5);
  for (int tile = blockIdx.x * (blockDim.x >> 5) + wave; tile < nTiles; tile += wstride) {
    const int nd = (tile << 5) + lane;   // one lane per node row
    if (nd < N) {  // stage A1 = [x, agg]
      const float* sX = x + (size_t)nd * DF;
      const float* sG = agg + (size_t)nd * DF;
      unsigned short* dst = sA1 + lane * 264;
#pragma unroll
      for (int c = 0; c < 128; c += 8) {
        *(uint4*)(dst + c)       = pk8(ld4(sX + c), ld4(sX + c + 4));
        *(uint4*)(dst + 128 + c) = pk8(ld4(sG + c), ld4(sG + c + 4));
      }
    }
    wait_lds();

    v8f acc0[8], acc1[8];
    zero_acc(acc0); zero_acc(acc1);
    gemm_tile2<256, 264>(sA1, sW1, lane, acc0, acc1);
    bias_relu_store2(acc0, acc1, b1, sA2, 136, lane);
    wait_lds();
    zero_acc(acc0); zero_acc(acc1);
    gemm_tile2<128, 136>(sA2, sW2, lane, acc0, acc1);
    bias_relu_store2(acc0, acc1, b2, sA3, 136, lane);
    wait_lds();
    zero_acc(acc0); zero_acc(acc1);
    gemm_tile2<128, 136>(sA3, sW3, lane, acc0, acc1);
    bias_store2_f32(acc0, acc1, b3, sLN, 132, lane);
    wait_lds();

    if (nd < N) {
      const float* hrow = sLN + lane * 132;
      float s = 0.f;
      for (int j = 0; j < 128; j += 4) {
        float4 h = *(const float4*)(hrow + j);
        s += h.x + h.y + h.z + h.w;
      }
      const float mu = s * (1.0f / 128.0f);
      float vs = 0.f;
      for (int j = 0; j < 128; j += 4) {
        float4 h = *(const float4*)(hrow + j);
        float a0 = h.x - mu, a1 = h.y - mu, a2 = h.z - mu, a3 = h.w - mu;
        vs += a0 * a0 + a1 * a1 + a2 * a2 + a3 * a3;
      }
      const float rstd = rsqrtf(vs * (1.0f / 128.0f) + 1e-5f);
      float* xRow = x + (size_t)nd * DF;
      for (int j = 0; j < 128; j += 4) {
        float4 h = *(const float4*)(hrow + j);
        float4 g4 = ld4(gg + j);
        float4 b4 = ld4(bt + j);
        float4 xo = ld4(xRow + j);
        float4 o;
        o.x = xo.x + (h.x - mu) * rstd * g4.x + b4.x;
        o.y = xo.y + (h.y - mu) * rstd * g4.y + b4.y;
        o.z = xo.z + (h.z - mu) * rstd * g4.z + b4.z;
        o.w = xo.w + (h.w - mu) * rstd * g4.w + b4.w;
        *(float4*)(xRow + j) = o;
      }
    }
  }
}

// ---------------- weight pre-pack: fp32 [K x 128] -> bf16 transposed [128 x (K+8)] (padded)
#define PER_L 153600
__global__ void pack_weights(const float* __restrict__ eW1, const float* __restrict__ eW2,
                             const float* __restrict__ eW3, const float* __restrict__ nW1,
                             const float* __restrict__ nW2, const float* __restrict__ nW3,
                             unsigned short* __restrict__ out) {
  const int total = 15 * PER_L;
  for (int idx = blockIdx.x * blockDim.x + threadIdx.x; idx < total;
       idx += gridDim.x * blockDim.x) {
    const int l = idx / PER_L, rr0 = idx % PER_L;
    const float* src; int K, S, base;
    if (rr0 < 50176)       { src = eW1 + l * 49152; K = 384; S = 392; base = 0; }
    else if (rr0 < 67584)  { src = eW2 + l * 16384; K = 128; S = 136; base = 50176; }
    else if (rr0 < 84992)  { src = eW3 + l * 16384; K = 128; S = 136; base = 67584; }
    else if (rr0 < 118784) { src = nW1 + l * 32768; K = 256; S = 264; base = 84992; }
    else if (rr0 < 136192) { src = nW2 + l * 16384; K = 128; S = 136; base = 118784; }
    else                   { src = nW3 + l * 16384; K = 128; S = 136; base = 136192; }
    const int rr = rr0 - base;
    const int n = rr / S, k = rr % S;
    out[idx] = (k < K) ? f2bf(src[k * DF + n]) : (unsigned short)0;
  }
}

__global__ void zero_f32(float* p, int n) {
  for (int i = blockIdx.x * blockDim.x + threadIdx.x; i < n; i += gridDim.x * blockDim.x)
    p[i] = 0.f;
}

extern "C" void kernel_launch(void* const* d_in, const int* in_sizes, int n_in,
                              void* d_out, int out_size, void* d_ws, size_t ws_size,
                              hipStream_t stream) {
  const float* x   = (const float*)d_in[0];
  const int*   ei  = (const int*)d_in[1];
  const float* ea  = (const float*)d_in[2];
  const float* eW1 = (const float*)d_in[3];
  const float* eb1 = (const float*)d_in[4];
  const float* eW2 = (const float*)d_in[5];
  const float* eb2 = (const float*)d_in[6];
  const float* eW3 = (const float*)d_in[7];
  const float* eb3 = (const float*)d_in[8];
  const float* eg  = (const float*)d_in[9];
  const float* ebt = (const float*)d_in[10];
  const float* nW1 = (const float*)d_in[11];
  const float* nb1 = (const float*)d_in[12];
  const float* nW2 = (const float*)d_in[13];
  const float* nb2 = (const float*)d_in[14];
  const float* nW3 = (const float*)d_in[15];
  const float* nb3 = (const float*)d_in[16];
  const float* ng  = (const float*)d_in[17];
  const float* nbt = (const float*)d_in[18];

  const int ND = in_sizes[0];         // N*D
  const int ED = in_sizes[2];         // E*D
  const int N = ND / DF, E = ED / DF;

  float* xOut  = (float*)d_out;               // working/final x
  float* eaOut = xOut + (size_t)ND;           // working/final edge_attr
  float* agg   = (float*)d_ws;                // N*D scatter buffer
  unsigned short* wpack = (unsigned short*)((char*)d_ws + (size_t)ND * sizeof(float));

  hipMemcpyAsync(xOut, x, (size_t)ND * sizeof(float), hipMemcpyDeviceToDevice, stream);
  hipMemcpyAsync(eaOut, ea, (size_t)ED * sizeof(float), hipMemcpyDeviceToDevice, stream);
  pack_weights<<<512, 256, 0, stream>>>(eW1, eW2, eW3, nW1, nW2, nW3, wpack);

  const int* rows = ei;
  const int* cols = ei + E;
  const size_t edgeLds = 2 * EW_LDS + 4 * (size_t)WV_BYTES;  // 307200
  const size_t nodeLds = 2 * NW_LDS + 4 * (size_t)WV_BYTES;  // 274432
  for (int l = 0; l < 15; ++l) {
    const unsigned short* wl = wpack + (size_t)l * PER_L;
    zero_f32<<<1024, 256, 0, stream>>>(agg, ND);
    edge_kernel<<<2048, 128, edgeLds, stream>>>(
        xOut, rows, cols, eaOut, agg, wl,
        eb1 + l * DF, eb2 + l * DF, eb3 + l * DF, eg + l * DF, ebt + l * DF, E);
    node_kernel<<<512, 128, nodeLds, stream>>>(
        xOut, agg, wl + 84992,
        nb1 + l * DF, nb2 + l * DF, nb3 + l * DF, ng + l * DF, nbt + l * DF, N);
  }
}
